// TrajectoryS4dModel_18253611008307
// MI455X (gfx1250) — matrix operand order repaired and checked by reading, not compiled, not run on hardware
//
#include <hip/hip_runtime.h>
#include <stdint.h>

#define B_  32
#define L_  512
#define SD_ 17
#define AD_ 6
#define DM_ 384
#define NL_ 12
#define N2_ 32
#define LT_ (4*L_)      // 2048
#define OC_ (2*DM_)     // 768
#define EPS_ 1e-5f

typedef __attribute__((ext_vector_type(16))) __bf16 v16bf;
typedef __attribute__((ext_vector_type(8)))  float  v8f;

union FragB16 { v16bf v; uint4 q[2]; };

__device__ __forceinline__ unsigned short f2bf(float f) {
  unsigned int u = __float_as_uint(f);
  unsigned int r = u + 0x7fffu + ((u >> 16) & 1u);
  return (unsigned short)(r >> 16);
}

// ---- CDNA5 async global->LDS copy (16B per lane), tracked by ASYNCcnt -----
__device__ __forceinline__ void async_ld16(unsigned lds_off, unsigned goff,
                                           const void* sbase) {
  // GVS form: mem_addr = SADDR + VADDR(i32); LDS dest = wave LDS base + vdst
  asm volatile("global_load_async_to_lds_b128 %0, %1, %2"
               :: "v"(lds_off), "v"(goff), "s"(sbase) : "memory");
}
__device__ __forceinline__ void wait_async0() {
  asm volatile("s_wait_asynccnt 0" ::: "memory");
}

// ---------------------------------------------------------------------------
// 1) Precompute discretized S4D params: w = exp(dt*A), c = 2*C*(exp(dt*A)-1)/A
// ---------------------------------------------------------------------------
__global__ void s4d_params_kernel(const float* __restrict__ log_dt,
                                  const float* __restrict__ C_ri,
                                  const float* __restrict__ log_A_real,
                                  const float* __restrict__ A_imag,
                                  float* __restrict__ wr_, float* __restrict__ wi_,
                                  float* __restrict__ cr_, float* __restrict__ ci_) {
  int idx = blockIdx.x * blockDim.x + threadIdx.x;
  if (idx >= NL_ * DM_ * N2_) return;
  int h  = (idx / N2_) % DM_;
  int li = idx / (N2_ * DM_);
  float dt = expf(log_dt[li * DM_ + h]);
  float Ar = -expf(log_A_real[idx]);
  float Ai = A_imag[idx];
  float dr = Ar * dt, di = Ai * dt;
  float e  = expf(dr);
  float wr = e * cosf(di), wi = e * sinf(di);
  float nr = wr - 1.f, ni = wi;
  float den = Ar * Ar + Ai * Ai;
  float qr = (nr * Ar + ni * Ai) / den;
  float qi = (ni * Ar - nr * Ai) / den;
  float Cr = C_ri[(size_t)idx * 2 + 0], Ci = C_ri[(size_t)idx * 2 + 1];
  wr_[idx] = wr; wi_[idx] = wi;
  cr_[idx] = 2.f * (Cr * qr - Ci * qi);
  ci_[idx] = 2.f * (Cr * qi + Ci * qr);
}

// ---------------------------------------------------------------------------
// 2) fp32 -> bf16 conversion (conv_w)
// ---------------------------------------------------------------------------
__global__ void cvt_bf16_kernel(const float* __restrict__ src,
                                unsigned short* __restrict__ dst, int n) {
  int i = blockIdx.x * blockDim.x + threadIdx.x;
  if (i < n) dst[i] = f2bf(src[i]);
}

// ---------------------------------------------------------------------------
// 3) Embedding + interleave: z (B, DM, 4L), slots (r,c,s,a)
// ---------------------------------------------------------------------------
__global__ void embed_kernel(const float* __restrict__ states,
                             const float* __restrict__ actions,
                             const float* __restrict__ rtg,
                             const float* __restrict__ ctg,
                             const int*   __restrict__ timesteps,
                             const float* __restrict__ W_es, const float* __restrict__ b_es,
                             const float* __restrict__ W_ea, const float* __restrict__ b_ea,
                             const float* __restrict__ W_er, const float* __restrict__ b_er,
                             const float* __restrict__ W_ec, const float* __restrict__ b_ec,
                             const float* __restrict__ E_t,
                             float* __restrict__ z) {
  int bl = blockIdx.x;            // b*L + l
  int h  = threadIdx.x;           // 0..383
  int b  = bl / L_, l = bl % L_;
  int ts = timesteps[bl];
  float te = E_t[(size_t)ts * DM_ + h];
  const float* st = states  + (size_t)bl * SD_;
  const float* ac = actions + (size_t)bl * AD_;
  float es = b_es[h] + te;
  #pragma unroll
  for (int j = 0; j < SD_; ++j) es += st[j] * W_es[j * DM_ + h];
  float ea = b_ea[h] + te;
  #pragma unroll
  for (int j = 0; j < AD_; ++j) ea += ac[j] * W_ea[j * DM_ + h];
  float er = b_er[h] + te + rtg[bl] * W_er[h];
  float ec = b_ec[h] + te + ctg[bl] * W_ec[h];
  float* zr = z + (size_t)(b * DM_ + h) * LT_ + 4 * l;
  zr[0] = er; zr[1] = ec; zr[2] = es; zr[3] = ea;
}

// ---------------------------------------------------------------------------
// 4) S4D diagonal recurrence scan: one wave per (b,h); lane = mode (N2 == 32).
//    Block covers 8 consecutive h of one batch; per 32-step chunk the 8 waves
//    transpose through a 32x8 LDS tile so the output is written as
//    yT (B, Lt, DM) bf16 in 16B rows -> the GEMM's B operand needs no
//    transpose and can be DMA'd straight into LDS.
// ---------------------------------------------------------------------------
__global__ __launch_bounds__(256)
void s4d_scan_kernel(const float* __restrict__ z,
                     const float* __restrict__ wr_, const float* __restrict__ wi_,
                     const float* __restrict__ cr_, const float* __restrict__ ci_,
                     const float* __restrict__ D_skip,
                     unsigned short* __restrict__ yT, int li) {
  __shared__ __align__(16) unsigned short lt_[32][8];
  int tid  = threadIdx.x;
  int lane = tid & 31, wv = tid >> 5;
  int blk  = blockIdx.x;                 // B * (DM/8)
  int b    = blk / (DM_ / 8);
  int h0   = (blk % (DM_ / 8)) * 8;
  int h    = h0 + wv;
  int pidx = (li * DM_ + h) * N2_ + lane;
  float wr = wr_[pidx], wi = wi_[pidx], cr = cr_[pidx], ci = ci_[pidx];
  float dsk = D_skip[li * DM_ + h];
  const float* zrow = z + (size_t)(b * DM_ + h) * LT_;
  float xr = 0.f, xi = 0.f;
  for (int l0 = 0; l0 < LT_; l0 += 32) {
    float uvec = zrow[l0 + lane];                       // coalesced 128B/wave
    if (l0 + 32 < LT_)
      __builtin_prefetch(zrow + l0 + 32 + lane, 0, 1);  // global_prefetch_b8
    float ysave = 0.f;
    #pragma unroll
    for (int j = 0; j < 32; ++j) {
      float u   = __shfl(uvec, j, 32);                  // broadcast u[l0+j]
      float nxr = wr * xr - wi * xi + u;                // x = w*x + u
      float nxi = wr * xi + wi * xr;
      xr = nxr; xi = nxi;
      float yn = cr * xr - ci * xi;                     // Re(c*x), 2x in c
      #pragma unroll
      for (int off = 16; off > 0; off >>= 1)
        yn += __shfl_xor(yn, off, 32);                  // sum over 32 modes
      if (lane == j) ysave = yn;
    }
    float s  = ysave + uvec * dsk;
    float ge = 0.5f * s * (1.f + erff(s * 0.70710678118654752f)); // exact GELU
    lt_[lane][wv] = f2bf(ge);
    __syncthreads();
    if (tid < 32) {
      uint4 v = *(const uint4*)&lt_[tid][0];
      *(uint4*)(yT + ((size_t)b * LT_ + l0 + tid) * DM_ + h0) = v;
    }
    __syncthreads();
  }
}

// ---------------------------------------------------------------------------
// 5) GLU channel-mix GEMM via WMMA bf16 (f32 accumulate).
//    C(768 x 2048) = conv_w(768x384) x Y(384x2048) per batch.
//    Y K-tiles are staged into LDS with double-buffered
//    global_load_async_to_lds_b128 (ASYNCcnt), overlapping DMA with WMMA.
//    Each wave owns both the a-half (rows m) and g-half (rows m+384) so
//    GLU + residual fuse in the epilogue.
// ---------------------------------------------------------------------------
#define GPITCH 40   // LDS row pitch (ushorts): 80B rows keep b128 loads aligned
__global__ __launch_bounds__(256)
void glu_gemm_kernel(const unsigned short* __restrict__ ybf,
                     const unsigned short* __restrict__ wbf,
                     const float* __restrict__ conv_b,
                     const float* __restrict__ z,
                     float* __restrict__ z2, int li) {
  __shared__ __align__(16) unsigned short ldsY[2][128 * GPITCH];
  int tid = threadIdx.x;
  int lane = tid & 31, w = tid >> 5;
  int wm = w & 3, wn = w >> 2;
  int b  = blockIdx.z;
  int n0 = blockIdx.x * 128;                  // l tile
  int m0 = blockIdx.y * 64 + wm * 16;         // a rows; g rows = m0 + 384
  int nw = n0 + wn * 64;

  v8f accA[4], accG[4];
  #pragma unroll
  for (int f = 0; f < 4; ++f)
    #pragma unroll
    for (int j = 0; j < 8; ++j) { accA[f][j] = 0.f; accG[f][j] = 0.f; }

  int rowA = m0 + (lane & 15);
  int rowG = m0 + DM_ + (lane & 15);
  int selA = (lane >> 4) * 8;                 // A frag: K half-select per ISA
  int selB = (lane >> 4) * 8;                 // B frag: K half-select per ISA
  const unsigned short* wbase = wbf + (size_t)li * OC_ * DM_;
  const unsigned short* ybase = ybf + (size_t)b * LT_ * DM_;   // yT (l, h)

  // per-thread DMA chunk coords: 512 x 16B chunks per tile, 2 per thread
  int c0 = tid, c1 = tid + 256;
  int l0r = c0 >> 2, p0 = c0 & 3;
  int l1r = c1 >> 2, p1 = c1 & 3;

  // prologue: DMA tile k=0 into buffer 0
  {
    unsigned g0 = (unsigned)(((n0 + l0r) * DM_) * 2 + p0 * 16);
    unsigned g1 = (unsigned)(((n0 + l1r) * DM_) * 2 + p1 * 16);
    async_ld16((unsigned)(uintptr_t)(&ldsY[0][0] + l0r * GPITCH + p0 * 8), g0, ybase);
    async_ld16((unsigned)(uintptr_t)(&ldsY[0][0] + l1r * GPITCH + p1 * 8), g1, ybase);
  }
  wait_async0();
  __syncthreads();

  for (int kk = 0; kk < DM_ / 32; ++kk) {
    int k0 = kk * 32;
    // prefetch next K tile into the other buffer (overlaps with WMMA below)
    if (kk + 1 < DM_ / 32) {
      unsigned short* nb = &ldsY[(kk + 1) & 1][0];
      unsigned g0 = (unsigned)(((n0 + l0r) * DM_ + k0 + 32) * 2 + p0 * 16);
      unsigned g1 = (unsigned)(((n0 + l1r) * DM_ + k0 + 32) * 2 + p1 * 16);
      async_ld16((unsigned)(uintptr_t)(nb + l0r * GPITCH + p0 * 8), g0, ybase);
      async_ld16((unsigned)(uintptr_t)(nb + l1r * GPITCH + p1 * 8), g1, ybase);
    }

    const unsigned short* cur = &ldsY[kk & 1][0];

    FragB16 fa, fg;
    const unsigned short* pa = wbase + (size_t)rowA * DM_ + k0 + selA;
    fa.q[0] = *(const uint4*)(pa);            // K {0..7} / {8..15}
    fa.q[1] = *(const uint4*)(pa + 16);       // K {16..23} / {24..31}
    const unsigned short* pg = wbase + (size_t)rowG * DM_ + k0 + selA;
    fg.q[0] = *(const uint4*)(pg);
    fg.q[1] = *(const uint4*)(pg + 16);

    #pragma unroll
    for (int f = 0; f < 4; ++f) {
      FragB16 fb;
      const unsigned short* pb = cur + (wn * 64 + f * 16 + (lane & 15)) * GPITCH + selB;
      fb.q[0] = *(const uint4*)(pb);          // K 0..7  (or 8..15)
      fb.q[1] = *(const uint4*)(pb + 16);     // K 16..23 (or 24..31)
      accA[f] = __builtin_amdgcn_wmma_f32_16x16x32_bf16(
          false, fa.v, false, fb.v, (short)0, accA[f], false, false);
      accG[f] = __builtin_amdgcn_wmma_f32_16x16x32_bf16(
          false, fg.v, false, fb.v, (short)0, accG[f], false, false);
    }

    wait_async0();        // next tile's DMA done (overlapped with the WMMAs)
    __syncthreads();
  }

  // epilogue: GLU + residual, write z2 fp32
  int mrow = (lane >> 4) * 8;
  #pragma unroll
  for (int f = 0; f < 4; ++f) {
    int l = nw + f * 16 + (lane & 15);
    #pragma unroll
    for (int j = 0; j < 8; ++j) {
      int m = m0 + mrow + j;
      float a  = accA[f][j] + conv_b[li * OC_ + m];
      float g  = accG[f][j] + conv_b[li * OC_ + m + DM_];
      float zo = z[(size_t)(b * DM_ + m) * LT_ + l];
      float sg = 1.f / (1.f + expf(-g));
      z2[(size_t)(b * DM_ + m) * LT_ + l] = a * sg + zo;
    }
  }
}

// ---------------------------------------------------------------------------
// 6) LayerNorm over DM per (b,l): tiled 32 l's per block for coalesced rows
// ---------------------------------------------------------------------------
__global__ __launch_bounds__(256)
void ln_kernel(const float* __restrict__ z2, float* __restrict__ z,
               const float* __restrict__ ln_g, const float* __restrict__ ln_b, int li) {
  __shared__ float sbuf[8][32], qbuf[8][32], mbuf[32], rbuf[32];
  int tid = threadIdx.x, lane = tid & 31, g = tid >> 5;
  int blk = blockIdx.x;
  int b  = blk >> 6;                 // Lt/32 = 64 tiles per batch
  int lt = (blk & 63) * 32;
  float vals[48];
  float s = 0.f, s2 = 0.f;
  #pragma unroll
  for (int r = 0; r < 48; ++r) {
    int h = g * 48 + r;
    float v = z2[(size_t)(b * DM_ + h) * LT_ + lt + lane];
    vals[r] = v; s += v; s2 += v * v;
  }
  sbuf[g][lane] = s; qbuf[g][lane] = s2;
  __syncthreads();
  if (g == 0) {
    float ts = 0.f, tq = 0.f;
    #pragma unroll
    for (int i = 0; i < 8; ++i) { ts += sbuf[i][lane]; tq += qbuf[i][lane]; }
    float mean = ts * (1.f / DM_);
    float var  = tq * (1.f / DM_) - mean * mean;
    mbuf[lane] = mean;
    rbuf[lane] = rsqrtf(var + EPS_);
  }
  __syncthreads();
  float mean = mbuf[lane], rstd = rbuf[lane];
  #pragma unroll
  for (int r = 0; r < 48; ++r) {
    int h = g * 48 + r;
    float v = (vals[r] - mean) * rstd * ln_g[li * DM_ + h] + ln_b[li * DM_ + h];
    z[(size_t)(b * DM_ + h) * LT_ + lt + lane] = v;
  }
}

// ---------------------------------------------------------------------------
// 7) Output heads: state_preds from slot-3 tokens @ W_ps, action_preds from
//    slot-2 tokens @ W_pa. LDS-tiled channel gather; tiny GEMMs.
// ---------------------------------------------------------------------------
__global__ __launch_bounds__(256)
void heads_kernel(const float* __restrict__ z,
                  const float* __restrict__ W_pa, const float* __restrict__ b_pa,
                  const float* __restrict__ W_ps, const float* __restrict__ b_ps,
                  float* __restrict__ out) {
  __shared__ float t2[96][32], t3[96][32];
  int tid = threadIdx.x, lane = tid & 31, g = tid >> 5;
  int blk = blockIdx.x;
  int b  = blk >> 4;                  // L/32 = 16 tiles per batch
  int lb = (blk & 15) * 32;
  int l  = lb + lane;
  float acc[3] = {0.f, 0.f, 0.f};
  for (int hc = 0; hc < 4; ++hc) {
    __syncthreads();
    #pragma unroll
    for (int i = 0; i < 12; ++i) {
      int rr = g + 8 * i;
      int h  = hc * 96 + rr;
      const float* zp = z + (size_t)(b * DM_ + h) * LT_ + 4 * l;
      t2[rr][lane] = zp[2];
      t3[rr][lane] = zp[3];
    }
    __syncthreads();
    for (int rr = 0; rr < 96; ++rr) {
      int h = hc * 96 + rr;
      float v2 = t2[rr][lane], v3 = t3[rr][lane];
      #pragma unroll
      for (int k = 0; k < 3; ++k) {
        int j = g + 8 * k;
        if (j < SD_)            acc[k] += v3 * W_ps[h * SD_ + j];
        else if (j < SD_ + AD_) acc[k] += v2 * W_pa[h * AD_ + (j - SD_)];
      }
    }
  }
  #pragma unroll
  for (int k = 0; k < 3; ++k) {
    int j = g + 8 * k;
    if (j < SD_) {
      out[((size_t)b * L_ + l) * SD_ + j] = acc[k] + b_ps[j];
    } else if (j < SD_ + AD_) {
      int jj = j - SD_;
      out[(size_t)B_ * L_ * SD_ + ((size_t)b * L_ + l) * AD_ + jj] = acc[k] + b_pa[jj];
    }
  }
}

// ---------------------------------------------------------------------------
// Host-side launch orchestration
// ---------------------------------------------------------------------------
extern "C" void kernel_launch(void* const* d_in, const int* in_sizes, int n_in,
                              void* d_out, int out_size, void* d_ws, size_t ws_size,
                              hipStream_t stream) {
  (void)in_sizes; (void)n_in; (void)out_size; (void)ws_size;

  const float* states     = (const float*)d_in[0];
  const float* actions    = (const float*)d_in[1];
  const float* rtg        = (const float*)d_in[2];
  const float* ctg        = (const float*)d_in[3];
  const int*   timesteps  = (const int*)  d_in[4];
  const float* W_es = (const float*)d_in[5];  const float* b_es = (const float*)d_in[6];
  const float* W_ea = (const float*)d_in[7];  const float* b_ea = (const float*)d_in[8];
  const float* W_er = (const float*)d_in[9];  const float* b_er = (const float*)d_in[10];
  const float* W_ec = (const float*)d_in[11]; const float* b_ec = (const float*)d_in[12];
  const float* E_t        = (const float*)d_in[13];
  const float* log_dt     = (const float*)d_in[14];
  const float* C_ri       = (const float*)d_in[15];
  const float* log_A_real = (const float*)d_in[16];
  const float* A_imag     = (const float*)d_in[17];
  const float* D_skip     = (const float*)d_in[18];
  const float* conv_w     = (const float*)d_in[19];
  const float* conv_b     = (const float*)d_in[20];
  const float* ln_g       = (const float*)d_in[21];
  const float* ln_b       = (const float*)d_in[22];
  const float* W_pa = (const float*)d_in[23]; const float* b_pa = (const float*)d_in[24];
  const float* W_ps = (const float*)d_in[25]; const float* b_ps = (const float*)d_in[26];

  // workspace layout (bytes)
  const size_t ZB   = (size_t)B_ * DM_ * LT_ * 4;        // 100,663,296
  const size_t YBFB = (size_t)B_ * DM_ * LT_ * 2;        //  50,331,648
  const size_t PN   = (size_t)NL_ * DM_ * N2_;           //     147,456 floats each
  char* ws = (char*)d_ws;
  float*          zbuf = (float*)(ws);
  float*          z2   = (float*)(ws + ZB);
  unsigned short* ybf  = (unsigned short*)(ws + 2 * ZB);   // yT (B, Lt, DM)
  float*          wr   = (float*)(ws + 2 * ZB + YBFB);
  float*          wi   = wr + PN;
  float*          cr   = wi + PN;
  float*          ci   = cr + PN;
  unsigned short* cwbf = (unsigned short*)(ws + 2 * ZB + YBFB + 4 * PN * 4);

  {
    int n = NL_ * DM_ * N2_;
    s4d_params_kernel<<<(n + 255) / 256, 256, 0, stream>>>(
        log_dt, C_ri, log_A_real, A_imag, wr, wi, cr, ci);
  }
  {
    int n = NL_ * OC_ * DM_;
    cvt_bf16_kernel<<<(n + 255) / 256, 256, 0, stream>>>(conv_w, cwbf, n);
  }
  embed_kernel<<<B_ * L_, DM_, 0, stream>>>(
      states, actions, rtg, ctg, timesteps,
      W_es, b_es, W_ea, b_ea, W_er, b_er, W_ec, b_ec, E_t, zbuf);

  for (int li = 0; li < NL_; ++li) {
    s4d_scan_kernel<<<B_ * (DM_ / 8), 256, 0, stream>>>(
        zbuf, wr, wi, cr, ci, D_skip, ybf, li);
    dim3 gg(LT_ / 128, DM_ / 64, B_);   // (16, 6, 32)
    glu_gemm_kernel<<<gg, 256, 0, stream>>>(ybf, cwbf, conv_b, zbuf, z2, li);
    ln_kernel<<<B_ * (LT_ / 32), 256, 0, stream>>>(z2, zbuf, ln_g, ln_b, li);
  }

  heads_kernel<<<B_ * (L_ / 32), 256, 0, stream>>>(
      zbuf, W_pa, b_pa, W_ps, b_ps, (float*)d_out);
}
